// CircuitGNN_57629871178420
// MI455X (gfx1250) — matrix operand order repaired and checked by reading, not compiled, not run on hardware
//
#include <hip/hip_runtime.h>

typedef __attribute__((ext_vector_type(16))) _Float16 v16h;
typedef __attribute__((ext_vector_type(8)))  float    v8f;

#define LOGIT 4
#define HID   16
#define DDIM  20   // LOGIT + HID

static __device__ __forceinline__ v8f wmma_f16(v16h a, v16h b, v8f c) {
  // D(16x16 f32) = A(16x32 f16) * B(32x16 f16) + C
  return __builtin_amdgcn_wmma_f32_16x16x32_f16(
      /*neg_a=*/false, a, /*neg_b=*/false, b,
      /*c_mod=*/(short)0, c, /*reuse_a=*/false, /*reuse_b=*/false);
}

// Load a 16x32 A-operand tile of W^T where W is row-major [Krows x Mcols]
// (W[k][m] at W[k*ldm+m]).  A layout used here (16-bit A 16x32, contiguous k):
//   lane L: row m = mbase + (L&15); element e: k = 16*(L>>4) + e
static __device__ __forceinline__ v16h load_wT_A(const float* __restrict__ W, int ldm,
                                                 int Krows, int Mcols,
                                                 int mbase, int kchunk, int lane) {
  const int m = mbase + (lane & 15);
  const int kbase = kchunk * 32 + ((lane >> 4) << 4);
  v16h a;
#pragma unroll
  for (int e = 0; e < 16; ++e) {
    const int k = kbase + e;
    float v = 0.0f;
    if (k < Krows && m < Mcols) v = W[k * ldm + m];
    a[e] = (_Float16)v;
  }
  return a;
}

// Bias tile as the WMMA C operand: C[m][n] = bias[m] for every column n.
// D layout: lane L, VGPR r -> row m = mbase + r + 8*(L>>4).
static __device__ __forceinline__ v8f load_bias_C(const float* __restrict__ bias,
                                                  int mbase, int mrows, int lane) {
  const int m0 = mbase + ((lane >> 4) << 3);
  v8f c;
#pragma unroll
  for (int r = 0; r < 8; ++r) {
    const int m = m0 + r;
    c[r] = (m < mrows) ? bias[m] : 0.0f;
  }
  return c;
}

static __device__ __forceinline__ v8f relu8(v8f d) {
#pragma unroll
  for (int r = 0; r < 8; ++r) d[r] = fmaxf(d[r], 0.0f);
  return d;
}

static __device__ __forceinline__ v8f xor16(v8f x) {
  v8f y;
#pragma unroll
  for (int r = 0; r < 8; ++r) y[r] = __shfl_xor(x[r], 16, 32);
  return y;
}

// Build the 32x16 f16 B operand (K = 0..31) of the next layer from two f32 D
// tiles (rows 0..15 in t0, rows 16..31 in t1).  B layout (dense 32x16, contiguous k
// order, as in load_wT_A): lane L holds column N=L&15, element e -> K = 16*(L>>4)+e.
// D layout: lane L holds column N=L&15, VGPR r -> row = r + 8*(L>>4), so each
// lane swaps one v8f half with its xor-16 partner lane.
static __device__ __forceinline__ v16h pack_B(v8f t0, v8f t1, int lane) {
  const bool hi = (lane & 16) != 0;
  const v8f t0p = xor16(t0);
  const v8f t1p = xor16(t1);
  v16h b;
#pragma unroll
  for (int r = 0; r < 8; ++r) {
    const float lo = hi ? t1p[r] : t0[r];   // K = 16*hi + r
    const float up = hi ? t1[r]  : t0p[r];  // K = 16*hi + 8 + r
    b[r]     = (_Float16)lo;
    b[8 + r] = (_Float16)up;
  }
  return b;
}

__global__ void __launch_bounds__(256) zero_f32(float* __restrict__ p, long n) {
  const long i = (long)blockIdx.x * blockDim.x + threadIdx.x;
  if (i < n) p[i] = 0.0f;
}

// Fused edge MLP + atomic scatter into agg[N x 20].  One wave = 16 edges.
__global__ void __launch_bounds__(256) edge_kernel(
    const float* __restrict__ logits, const float* __restrict__ hidden,
    const int* __restrict__ senders, const int* __restrict__ receivers,
    const float* __restrict__ W1, const float* __restrict__ b1,
    const float* __restrict__ W2, const float* __restrict__ b2,
    const float* __restrict__ W3, const float* __restrict__ b3,
    float* __restrict__ agg, int E) {
  const int lane  = threadIdx.x & 31;
  const int wave  = threadIdx.x >> 5;
  const int tile  = blockIdx.x * (blockDim.x >> 5) + wave;
  const int ebase = tile * 16;
  if (ebase >= E) return;  // wave-uniform: EXEC stays all-ones for WMMA

  // Weight A operands (transposed weights) + bias C tiles; all L2-resident.
  v16h a1[4], a2[2][2], a3[2];
  v8f  c1[4], c2[2], c3[2];
#pragma unroll
  for (int t = 0; t < 4; ++t) {
    a1[t] = load_wT_A(W1, 64, DDIM, 64, 16 * t, 0, lane);
    c1[t] = load_bias_C(b1, 16 * t, 64, lane);
  }
#pragma unroll
  for (int t = 0; t < 2; ++t) {
#pragma unroll
    for (int c = 0; c < 2; ++c) a2[t][c] = load_wT_A(W2, 32, 64, 32, 16 * t, c, lane);
    c2[t] = load_bias_C(b2, 16 * t, 32, lane);
    a3[t] = load_wT_A(W3, DDIM, 32, DDIM, 16 * t, 0, lane);
    c3[t] = load_bias_C(b3, 16 * t, DDIM, lane);
  }

  const int eidx   = ebase + (lane & 15);
  const int edge   = (eidx < E) ? eidx : (E - 1);
  const bool valid = (eidx < E);
  const int s = senders[edge];

  // B operand: X^T tile, K = feature (padded to 32), N = edge-in-tile.
  // element e -> k = 16*(lane>>4) + e.
  const int kb0 = (lane >> 4) << 4;
  v16h b0;
#pragma unroll
  for (int e = 0; e < 16; ++e) {
    const int k = kb0 + e;
    float v = 0.0f;
    if (k < LOGIT)      v = logits[s * LOGIT + k];
    else if (k < DDIM)  v = hidden[s * HID + (k - LOGIT)];
    b0[e] = (_Float16)v;
  }

  // Layer 1: Y1 = W1^T (64x20) * X^T (20x16) + b1
  v8f d1[4];
#pragma unroll
  for (int t = 0; t < 4; ++t) d1[t] = relu8(wmma_f16(a1[t], b0, c1[t]));
  const v16h h0 = pack_B(d1[0], d1[1], lane);  // K = 0..31
  const v16h h1 = pack_B(d1[2], d1[3], lane);  // K = 32..63

  // Layer 2: Y2 = W2^T (32x64) * Y1 (64x16) + b2
  v8f d2[2];
#pragma unroll
  for (int t = 0; t < 2; ++t) {
    d2[t] = wmma_f16(a2[t][0], h0, c2[t]);
    d2[t] = relu8(wmma_f16(a2[t][1], h1, d2[t]));
  }
  const v16h g0 = pack_B(d2[0], d2[1], lane);  // K = 0..31

  // Layer 3: Y3 = W3^T (20x32) * Y2 (32x16) + b3 (no relu)
  v8f d3[2];
#pragma unroll
  for (int t = 0; t < 2; ++t) d3[t] = wmma_f16(a3[t], g0, c3[t]);

  // Scatter: agg[receiver][m] += msg[m][edge]; atomics stay L2-resident.
  const int rnode = receivers[edge];
  const int mhi = (lane >> 4) << 3;  // 0 or 8
  if (valid) {
#pragma unroll
    for (int r = 0; r < 8; ++r)
      atomicAdd(&agg[rnode * DDIM + (mhi + r)], d3[0][r]);
    if (lane < 16) {
#pragma unroll
      for (int r = 0; r < 4; ++r)
        atomicAdd(&agg[rnode * DDIM + 16 + r], d3[1][r]);
    }
  }
}

// Node MLP: in = [logits | hidden | agg] (40 features, padded to 64 -> 2 B chunks).
__global__ void __launch_bounds__(256) node_kernel(
    const float* __restrict__ logits, const float* __restrict__ hidden,
    const float* __restrict__ agg,
    const float* __restrict__ W1, const float* __restrict__ b1,
    const float* __restrict__ W2, const float* __restrict__ b2,
    const float* __restrict__ W3, const float* __restrict__ b3,
    float* __restrict__ out_logits, float* __restrict__ out_hidden, int N) {
  const int lane  = threadIdx.x & 31;
  const int wave  = threadIdx.x >> 5;
  const int tile  = blockIdx.x * (blockDim.x >> 5) + wave;
  const int nbase = tile * 16;
  if (nbase >= N) return;

  v16h a1[4][2], a2[2][2], a3[2];
  v8f  c1[4], c2[2], c3[2];
#pragma unroll
  for (int t = 0; t < 4; ++t) {
#pragma unroll
    for (int c = 0; c < 2; ++c) a1[t][c] = load_wT_A(W1, 64, 2 * DDIM, 64, 16 * t, c, lane);
    c1[t] = load_bias_C(b1, 16 * t, 64, lane);
  }
#pragma unroll
  for (int t = 0; t < 2; ++t) {
#pragma unroll
    for (int c = 0; c < 2; ++c) a2[t][c] = load_wT_A(W2, 32, 64, 32, 16 * t, c, lane);
    c2[t] = load_bias_C(b2, 16 * t, 32, lane);
    a3[t] = load_wT_A(W3, DDIM, 32, DDIM, 16 * t, 0, lane);
    c3[t] = load_bias_C(b3, 16 * t, DDIM, lane);
  }

  const int nidx   = nbase + (lane & 15);
  const int node   = (nidx < N) ? nidx : (N - 1);
  const bool valid = (nidx < N);

  const int kb0 = (lane >> 4) << 4;
  v16h bf[2];
#pragma unroll
  for (int c = 0; c < 2; ++c) {
#pragma unroll
    for (int e = 0; e < 16; ++e) {
      const int k = 32 * c + kb0 + e;
      float v = 0.0f;
      if (k < LOGIT)          v = logits[node * LOGIT + k];
      else if (k < DDIM)      v = hidden[node * HID + (k - LOGIT)];
      else if (k < 2 * DDIM)  v = agg[node * DDIM + (k - DDIM)];
      bf[c][e] = (_Float16)v;
    }
  }

  v8f d1[4];
#pragma unroll
  for (int t = 0; t < 4; ++t) {
    d1[t] = wmma_f16(a1[t][0], bf[0], c1[t]);
    d1[t] = relu8(wmma_f16(a1[t][1], bf[1], d1[t]));
  }
  const v16h h0 = pack_B(d1[0], d1[1], lane);
  const v16h h1 = pack_B(d1[2], d1[3], lane);

  v8f d2[2];
#pragma unroll
  for (int t = 0; t < 2; ++t) {
    d2[t] = wmma_f16(a2[t][0], h0, c2[t]);
    d2[t] = relu8(wmma_f16(a2[t][1], h1, d2[t]));
  }
  const v16h g0 = pack_B(d2[0], d2[1], lane);

  v8f d3[2];
#pragma unroll
  for (int t = 0; t < 2; ++t) d3[t] = wmma_f16(a3[t], g0, c3[t]);

  if (valid) {
    const int mhi = (lane >> 4) << 3;  // 0 or 8
#pragma unroll
    for (int r = 0; r < 8; ++r) {
      const int m = mhi + r;                 // 0..15
      const float v = d3[0][r];
      if (m < LOGIT) out_logits[node * LOGIT + m] = v;
      else           out_hidden[node * HID + (m - LOGIT)] = v;
    }
    if (lane < 16) {
#pragma unroll
      for (int r = 0; r < 4; ++r)            // m = 16..19 -> hidden 12..15
        out_hidden[node * HID + 12 + r] = d3[1][r];
    }
  }
}

extern "C" void kernel_launch(void* const* d_in, const int* in_sizes, int n_in,
                              void* d_out, int out_size, void* d_ws, size_t ws_size,
                              hipStream_t stream) {
  const float* logits    = (const float*)d_in[0];
  const float* hidden    = (const float*)d_in[1];
  const int*   senders   = (const int*)d_in[2];
  const int*   receivers = (const int*)d_in[3];
  const float* We1 = (const float*)d_in[4];
  const float* be1 = (const float*)d_in[5];
  const float* We2 = (const float*)d_in[6];
  const float* be2 = (const float*)d_in[7];
  const float* We3 = (const float*)d_in[8];
  const float* be3 = (const float*)d_in[9];
  const float* Wn1 = (const float*)d_in[10];
  const float* bn1 = (const float*)d_in[11];
  const float* Wn2 = (const float*)d_in[12];
  const float* bn2 = (const float*)d_in[13];
  const float* Wn3 = (const float*)d_in[14];
  const float* bn3 = (const float*)d_in[15];

  const int N = in_sizes[0] / LOGIT;
  const int E = in_sizes[2];

  float* agg = (float*)d_ws;                 // N x 20 scratch, zeroed each call
  float* out_logits = (float*)d_out;         // N x 4
  float* out_hidden = out_logits + (size_t)N * LOGIT;  // N x 16

  const long aggN = (long)N * DDIM;
  zero_f32<<<(int)((aggN + 255) / 256), 256, 0, stream>>>(agg, aggN);

  const int etiles = (E + 15) / 16;
  edge_kernel<<<(etiles + 7) / 8, 256, 0, stream>>>(
      logits, hidden, senders, receivers,
      We1, be1, We2, be2, We3, be3, agg, E);

  const int ntiles = (N + 15) / 16;
  node_kernel<<<(ntiles + 7) / 8, 256, 0, stream>>>(
      logits, hidden, agg,
      Wn1, bn1, Wn2, bn2, Wn3, bn3, out_logits, out_hidden, N);
}
